// AdversaryLayer_38723425140998
// MI455X (gfx1250) — matrix operand order repaired and checked by reading, not compiled, not run on hardware
//
#include <hip/hip_runtime.h>
#include <hip/hip_bf16.h>
#include <math.h>

// ---------------------------------------------------------------------------
// Problem constants: B=256 batch, S=512 steps, UNITS=256 hidden, Z_K=256.
// Key structural fact: all GEMMs reduce over units, never batch -> each
// 16-row batch strip is an independent recurrence -> one persistent block
// per strip, zero inter-block synchronization, state lives in LDS.
// ---------------------------------------------------------------------------
#define NB   256
#define NS   512
#define NU   256
#define NZK  256

typedef __attribute__((ext_vector_type(16))) __bf16 v16bf;
typedef __attribute__((ext_vector_type(8)))  __bf16 v8bf;
typedef __attribute__((ext_vector_type(8)))  float  v8f;

__device__ __forceinline__ float sigmoidf_(float x) {
    return 1.0f / (1.0f + expf(-x));
}

// ---------------------------------------------------------------------------
// A-fragment loader from a bf16 row-major buffer (LDS here; addrspace is
// inferred after inlining).
// Layout used here (16-bit A 16x32, contiguous k, same as B): lanes 0-15 -> K = kk..kk+15,
// lanes 16-31 -> K = kk+16..kk+31; row M = mrow.
// Two contiguous 16-byte loads, no converts.
// ---------------------------------------------------------------------------
__device__ __forceinline__ v16bf load_a_frag(const __bf16* A,
                                             int mrow, int kk, int hi) {
    const __bf16* pa = A + mrow * NU + kk + 16 * hi;
    v8bf lo = *(const v8bf*)(pa);
    v8bf hv = *(const v8bf*)(pa + 8);
    return __builtin_shufflevector(lo, hv, 0, 1, 2, 3, 4, 5, 6, 7,
                                           8, 9, 10, 11, 12, 13, 14, 15);
}

// ---------------------------------------------------------------------------
// B-fragment loader: weights pre-transposed to Wt[n][k] (bf16, 256x256).
// Layout used here (16-bit B 32x16, contiguous k, same as A): lane n = n0+(lane&15); lanes 0-15 hold
// K = kk..kk+15, lanes 16-31 hold K = kk+16..kk+31 -> one contiguous v16bf.
// ---------------------------------------------------------------------------
__device__ __forceinline__ v16bf load_b_frag(const __bf16* __restrict__ Wt,
                                             int ncol, int kk, int hi) {
    return *(const v16bf*)(Wt + ncol * NU + kk + 16 * hi);
}

// Full 16x16-tile GEMM over K=256 (8 WMMA ops), f32 accum. A is an LDS strip
// (16 rows x 256 cols, row-major); local row = lane&15.
__device__ __forceinline__ v8f gemm_tile(const __bf16* A,
                                         const __bf16* __restrict__ Wt,
                                         int n0, int lane) {
    const int hi = lane >> 4;
    const int ll = lane & 15;
    v8f acc = {};
#pragma unroll
    for (int kk = 0; kk < NU; kk += 32) {
        v16bf a = load_a_frag(A, ll, kk, hi);
        v16bf b = load_b_frag(Wt, n0 + ll, kk, hi);
        acc = __builtin_amdgcn_wmma_f32_16x16x32_bf16(
            false, a, false, b, (short)0, acc, false, false);
    }
    return acc;
}

// ---------------------------------------------------------------------------
// Prep: convert the 7 weight matrices (row-major [k][n], f32) to bf16,
// transposed to [n][k], packed consecutively in workspace.
// ---------------------------------------------------------------------------
__global__ void __launch_bounds__(256)
pack_weights_kernel(const float* __restrict__ hU,
                    const float* __restrict__ fW,
                    const float* __restrict__ iW,
                    const float* __restrict__ cW,
                    const float* __restrict__ oW,
                    const float* __restrict__ tW,
                    const float* __restrict__ yW,
                    __bf16* __restrict__ dst) {
    int idx = blockIdx.x * blockDim.x + threadIdx.x;   // 7 * 65536 total
    if (idx >= 7 * NU * NU) return;
    int w = idx >> 16;
    int e = idx & 0xFFFF;
    int k = e >> 8;
    int n = e & 255;
    const float* src;
    switch (w) {
        case 0: src = hU; break;
        case 1: src = fW; break;
        case 2: src = iW; break;
        case 3: src = cW; break;
        case 4: src = oW; break;
        case 5: src = tW; break;
        default: src = yW; break;
    }
    dst[w * NU * NU + n * NU + k] = (__bf16)src[k * NU + n];
}

// ---------------------------------------------------------------------------
// Persistent recurrence kernel: 16 blocks x 512 threads (16 waves).
// Block b owns batch rows [16b, 16b+16) for all 512 timesteps.
// All state in LDS (64 KB): hprev (f32+bf16), h, g, t strips, logits.
// Per step, per wave (one 16-col tile): 8+32+8+8 = 56 WMMA ops.
// __launch_bounds__(512, 1): one workgroup per WGP -> 4 waves/SIMD, so the
// register allocator may hold the loop-invariant h_U B-fragments in VGPRs
// across the step loop WITHOUT spilling to scratch.
// ---------------------------------------------------------------------------
__global__ void __launch_bounds__(512, 1)
rnn_persistent_kernel(const __bf16* __restrict__ wt,  // 7 packed matrices
                      const float* __restrict__ embed,// (Z_K+1, U)
                      const float* __restrict__ h_b,
                      const float* __restrict__ f_b,
                      const float* __restrict__ i_b,
                      const float* __restrict__ c_b,
                      const float* __restrict__ o_b,
                      const float* __restrict__ t_b,
                      const float* __restrict__ y_b,
                      const int* __restrict__ z,
                      const float* __restrict__ h0,
                      float* __restrict__ out) {
    __shared__ float  hprev_f[16 * NU];   // 16 KB  f32 state (elementwise)
    __shared__ __bf16 hprev_b[16 * NU];   //  8 KB  bf16 state (A operand)
    __shared__ __bf16 hstrip [16 * NU];   //  8 KB  tanh hidden
    __shared__ __bf16 gstrip [16 * NU];   //  8 KB  o*h1
    __shared__ __bf16 tstrip [16 * NU];   //  8 KB  tanh(t)
    __shared__ float  logits [16 * NZK];  // 16 KB  pre-softmax

    const __bf16* wt_hU = wt + 0 * NU * NU;
    const __bf16* wt_f  = wt + 1 * NU * NU;
    const __bf16* wt_i  = wt + 2 * NU * NU;
    const __bf16* wt_c  = wt + 3 * NU * NU;
    const __bf16* wt_o  = wt + 4 * NU * NU;
    const __bf16* wt_t  = wt + 5 * NU * NU;
    const __bf16* wt_y  = wt + 6 * NU * NU;

    const int lane = threadIdx.x & 31;
    const int wave = threadIdx.x >> 5;    // 0..15 -> n-tile / softmax row
    const int m0   = blockIdx.x * 16;     // batch-row strip
    const int n0   = wave * 16;
    const int hi   = lane >> 4;
    const int ll   = lane & 15;
    const int n    = n0 + ll;

    // Seed state strip from h0 (broadcast over batch rows).
    for (int idx = threadIdx.x; idx < 16 * NU; idx += blockDim.x) {
        float v = h0[idx & (NU - 1)];
        hprev_f[idx] = v;
        hprev_b[idx] = (__bf16)v;
    }
    __syncthreads();

    // Per-thread loop-invariant biases.
    const float hb = h_b[n];
    const float fb = f_b[n], ib = i_b[n], cb = c_b[n], ob = o_b[n];
    const float tb = t_b[n], yb = y_b[n];

    for (int s = 0; s < NS; ++s) {
        // ---- Phase 1: h = tanh(embed[z_shift] + h_prev @ h_U + h_b) ----
        {
            v8f acc = gemm_tile(hprev_b, wt_hU, n0, lane);
#pragma unroll
            for (int j = 0; j < 8; ++j) {
                int ml = j + 8 * hi;                  // local row 0..15
                int m  = m0 + ml;
                int code = (s == 0) ? 0 : (z[m * NS + (s - 1)] + 1);
                float v = acc[j] + embed[code * NU + n] + hb;
                hstrip[ml * NU + n] = (__bf16)tanhf(v);
            }
        }
        __syncthreads();

        // ---- Phase 2: gates (shared A), h1 = hprev*f + c*i, g = o*h1 ----
        {
            v8f af = {}, ai = {}, ac = {}, ao = {};
#pragma unroll
            for (int kk = 0; kk < NU; kk += 32) {
                v16bf a   = load_a_frag(hstrip, ll, kk, hi);
                v16bf bf_ = load_b_frag(wt_f, n, kk, hi);
                af = __builtin_amdgcn_wmma_f32_16x16x32_bf16(false, a, false, bf_, (short)0, af, false, false);
                v16bf bi_ = load_b_frag(wt_i, n, kk, hi);
                ai = __builtin_amdgcn_wmma_f32_16x16x32_bf16(false, a, false, bi_, (short)0, ai, false, false);
                v16bf bc_ = load_b_frag(wt_c, n, kk, hi);
                ac = __builtin_amdgcn_wmma_f32_16x16x32_bf16(false, a, false, bc_, (short)0, ac, false, false);
                v16bf bo_ = load_b_frag(wt_o, n, kk, hi);
                ao = __builtin_amdgcn_wmma_f32_16x16x32_bf16(false, a, false, bo_, (short)0, ao, false, false);
            }
#pragma unroll
            for (int j = 0; j < 8; ++j) {
                int ml = j + 8 * hi;
                float fg = sigmoidf_(af[j] + fb);
                float ig = sigmoidf_(ai[j] + ib);
                float cg = tanhf(ac[j] + cb);
                float og = sigmoidf_(ao[j] + ob);
                float hp = hprev_f[ml * NU + n];      // own slot: safe RMW
                float h1 = hp * fg + cg * ig;
                hprev_f[ml * NU + n] = h1;
                hprev_b[ml * NU + n] = (__bf16)h1;
                gstrip [ml * NU + n] = (__bf16)(og * h1);
            }
        }
        __syncthreads();

        // ---- Phase 3: t = tanh(g @ t_W + t_b) ----
        {
            v8f acc = gemm_tile(gstrip, wt_t, n0, lane);
#pragma unroll
            for (int j = 0; j < 8; ++j) {
                int ml = j + 8 * hi;
                tstrip[ml * NU + n] = (__bf16)tanhf(acc[j] + tb);
            }
        }
        __syncthreads();

        // ---- Phase 4: logits = t @ y_W + y_b ----
        {
            v8f acc = gemm_tile(tstrip, wt_y, n0, lane);
#pragma unroll
            for (int j = 0; j < 8; ++j) {
                int ml = j + 8 * hi;
                logits[ml * NZK + n] = acc[j] + yb;
            }
        }
        __syncthreads();

        // ---- Phase 5: row softmax (wave w = local row w) -> global out ----
        {
            float v[8];
            float mx = -1e30f;
#pragma unroll
            for (int e = 0; e < 8; ++e) {
                v[e] = logits[wave * NZK + lane + e * 32];
                mx = fmaxf(mx, v[e]);
            }
#pragma unroll
            for (int off = 16; off >= 1; off >>= 1)
                mx = fmaxf(mx, __shfl_xor(mx, off));

            float sum = 0.0f;
#pragma unroll
            for (int e = 0; e < 8; ++e) {
                v[e] = expf(v[e] - mx);
                sum += v[e];
            }
#pragma unroll
            for (int off = 16; off >= 1; off >>= 1)
                sum += __shfl_xor(sum, off);

            const float inv = 1.0f / sum;
            const long base = ((long)(m0 + wave) * NS + s) * NZK;
#pragma unroll
            for (int e = 0; e < 8; ++e)
                out[base + lane + e * 32] = v[e] * inv;
        }
        __syncthreads();
    }
}

// ---------------------------------------------------------------------------
// Host-side launch: 2 launches total (weight pack + persistent recurrence).
// Graph-capture safe; deterministic/idempotent every call.
// ---------------------------------------------------------------------------
extern "C" void kernel_launch(void* const* d_in, const int* in_sizes, int n_in,
                              void* d_out, int out_size, void* d_ws, size_t ws_size,
                              hipStream_t stream) {
    const int*   z    = (const int*)  d_in[0];
    const float* h_W  = (const float*)d_in[1];   // (Z_K+1, U)
    const float* h_U  = (const float*)d_in[2];
    const float* h_b  = (const float*)d_in[3];
    const float* f_W  = (const float*)d_in[4];
    const float* f_b  = (const float*)d_in[5];
    const float* i_W  = (const float*)d_in[6];
    const float* i_b  = (const float*)d_in[7];
    const float* c_W  = (const float*)d_in[8];
    const float* c_b  = (const float*)d_in[9];
    const float* o_W  = (const float*)d_in[10];
    const float* o_b  = (const float*)d_in[11];
    const float* t_W  = (const float*)d_in[12];
    const float* t_b  = (const float*)d_in[13];
    const float* y_W  = (const float*)d_in[14];
    const float* y_b  = (const float*)d_in[15];
    const float* h0   = (const float*)d_in[16];
    float* out = (float*)d_out;

    const size_t wbytes = (size_t)NU * NU * sizeof(__bf16);   // 128 KB / matrix
    if (ws_size < 7 * wbytes) return;
    __bf16* wt = (__bf16*)d_ws;                               // 7 packed matrices

    pack_weights_kernel<<<(7 * NU * NU + 255) / 256, 256, 0, stream>>>(
        h_U, f_W, i_W, c_W, o_W, t_W, y_W, wt);

    rnn_persistent_kernel<<<16, 512, 0, stream>>>(
        wt, h_W, h_b, f_b, i_b, c_b, o_b, t_b, y_b, z, h0, out);
}
